// MultiHeadAttention_51419348468282
// MI455X (gfx1250) — matrix operand order repaired and checked by reading, not compiled, not run on hardware
//
#include <hip/hip_runtime.h>
#include <hip/hip_bf16.h>

typedef __attribute__((ext_vector_type(16))) __bf16 v16bf;
typedef __attribute__((ext_vector_type(8)))  __bf16 v8bf;
typedef __attribute__((ext_vector_type(8)))  float  v8f;
typedef __attribute__((ext_vector_type(4)))  float  v4f;
typedef __attribute__((ext_vector_type(4)))  int    v4i;

#define WMMA_BF16(a, b, c) \
  __builtin_amdgcn_wmma_f32_16x16x32_bf16(false, (a), false, (b), (short)0, (c), false, false)

// Build a 16-element bf16 fragment from two 16-byte-aligned 8-element runs.
static __device__ inline v16bf make_frag(const __bf16* p0, const __bf16* p1) {
  v8bf lo = *(const v8bf*)p0;
  v8bf hi = *(const v8bf*)p1;
  v16bf r;
#pragma unroll
  for (int i = 0; i < 8; ++i) { r[i] = lo[i]; r[i + 8] = hi[i]; }
  return r;
}

// 16-byte global -> LDS copy.  Uses the gfx1250 async-to-LDS path (ASYNCcnt,
// no VGPR round trip) when the builtin is available; else a plain vector copy.
// Builtin signature (from compiler diagnostic): arg0 = AS1 (device) v4i*,
// arg1 = LDS v4i*, then imm offset and imm cpol.
static __device__ inline void async_copy16(const __bf16* g, __bf16* l) {
#if __has_builtin(__builtin_amdgcn_global_load_async_to_lds_b128)
  __builtin_amdgcn_global_load_async_to_lds_b128(
      (__attribute__((address_space(1))) v4i*)g,
      (__attribute__((address_space(3))) v4i*)l, 0, 0);
#else
  *(v8bf*)l = *(const v8bf*)g;
#endif
}

static __device__ inline void wait_async() {
#if __has_builtin(__builtin_amdgcn_s_wait_asynccnt)
  __builtin_amdgcn_s_wait_asynccnt(0);
#else
  asm volatile("s_wait_asynccnt 0" ::: "memory");
#endif
}

// ---------------------------------------------------------------------------
// Kernel 1: QKV projection.  q/k/v[b,h,s,dk] = X[b,s,:] . W[h,:,dk]  (bf16 out)
// Grid: (M/256, H, 3), block 256 (8 waves).  Tile: 256(M) x 64(N), K chunks 32.
// Each wave owns 32 M rows (2 A frags) -> 8 WMMAs per staged chunk.
// ---------------------------------------------------------------------------
__global__ __launch_bounds__(256)
void qkv_proj_kernel(const float* __restrict__ Qin, const float* __restrict__ Kin,
                     const float* __restrict__ Vin,
                     const float* __restrict__ Wq, const float* __restrict__ Wk,
                     const float* __restrict__ Wv,
                     __bf16* __restrict__ qo, __bf16* __restrict__ ko,
                     __bf16* __restrict__ vo) {
  constexpr int DM = 1024, DK = 64, S = 2048, H = 16;
  const int z = blockIdx.z;
  const float* X = (z == 0) ? Qin : (z == 1) ? Kin : Vin;
  const float* W = (z == 0) ? Wq : (z == 1) ? Wk : Wv;
  __bf16* O = (z == 0) ? qo : (z == 1) ? ko : vo;
  const int h = blockIdx.y;
  const int row0 = blockIdx.x * 256;
  const float* Wh = W + (size_t)h * DM * DK;

  __shared__ __align__(16) __bf16 lA[256][40];  // padded stride: 16B-aligned frags
  __shared__ __align__(16) __bf16 lB[64][40];   // B staged transposed: [n][k]

  const int tid = threadIdx.x;
  const int lane = tid & 31, wave = tid >> 5;
  const int m0 = wave * 32;
  const int nrow = lane & 15, half = lane >> 4;

  v8f acc[2][4] = {};

  for (int kk = 0; kk < DM; kk += 32) {
    // stage A: 256x32 f32 -> bf16 (coalesced float4 reads)
#pragma unroll
    for (int t = 0; t < 8; ++t) {
      int idx = tid + t * 256;
      int r = idx >> 3, c4 = idx & 7;
      v4f d = *(const v4f*)&X[(size_t)(row0 + r) * DM + kk + c4 * 4];
#pragma unroll
      for (int j = 0; j < 4; ++j) lA[r][c4 * 4 + j] = (__bf16)d[j];
    }
    // stage B transposed: W rows kk..kk+31, 64 cols
#pragma unroll
    for (int t = 0; t < 2; ++t) {
      int idx = tid + t * 256;
      int r = idx >> 4, c4 = idx & 15;
      v4f d = *(const v4f*)&Wh[(size_t)(kk + r) * DK + c4 * 4];
#pragma unroll
      for (int j = 0; j < 4; ++j) lB[c4 * 4 + j][r] = (__bf16)d[j];
    }
    __syncthreads();

    v16bf a[2];
#pragma unroll
    for (int mi = 0; mi < 2; ++mi)
      a[mi] = make_frag(&lA[m0 + mi * 16 + nrow][half * 8],
                        &lA[m0 + mi * 16 + nrow][16 + half * 8]);
#pragma unroll
    for (int nt = 0; nt < 4; ++nt) {
      v16bf b = make_frag(&lB[nt * 16 + nrow][half * 8],
                          &lB[nt * 16 + nrow][16 + half * 8]);
#pragma unroll
      for (int mi = 0; mi < 2; ++mi) acc[mi][nt] = WMMA_BF16(a[mi], b, acc[mi][nt]);
    }
    __syncthreads();
  }

  // C layout: lane holds rows m=(lane/16)*8+r, col n=lane%16
#pragma unroll
  for (int mi = 0; mi < 2; ++mi)
#pragma unroll
    for (int nt = 0; nt < 4; ++nt)
#pragma unroll
      for (int r = 0; r < 8; ++r) {
        int gr = row0 + m0 + mi * 16 + half * 8 + r;
        int b = gr >> 11, s = gr & 2047;  // S = 2048
        O[((size_t)(b * H + h) * S + s) * DK + nt * 16 + nrow] =
            (__bf16)acc[mi][nt][r];
      }
}

// ---------------------------------------------------------------------------
// Kernel 2: flash attention.  Each wave owns a 16-query-row tile; 8 waves share
// LDS-staged 32-key K/V chunks (K via async-to-LDS). Online softmax, P bounced
// through LDS (C layout -> A layout), then P x V WMMAs.
// Grid: (S/128, B*H), block 256.
// ---------------------------------------------------------------------------
__global__ __launch_bounds__(256)
void attention_kernel(const __bf16* __restrict__ q, const __bf16* __restrict__ k,
                      const __bf16* __restrict__ v, __bf16* __restrict__ heads) {
  constexpr int S = 2048, DK = 64, H = 16;
  const int bh = blockIdx.y;
  const int bb = bh >> 4, h = bh & 15;
  const __bf16* qb = q + (size_t)bh * S * DK;
  const __bf16* kb = k + (size_t)bh * S * DK;
  const __bf16* vb = v + (size_t)bh * S * DK;

  __shared__ __align__(16) __bf16 lK[32][72];      // [key][dk]
  __shared__ __align__(16) __bf16 lV[64][40];      // transposed: [dk][key]
  __shared__ __align__(16) __bf16 lP[8][16][40];   // per-wave P bounce

  const int tid = threadIdx.x, lane = tid & 31, wave = tid >> 5;
  const int nrow = lane & 15, half = lane >> 4;
  const int m0 = blockIdx.x * 128 + wave * 16;

  // Load the 16x64 query tile once as two K=32 A-fragments.
  v16bf qf[2];
#pragma unroll
  for (int ks = 0; ks < 2; ++ks)
    qf[ks] = make_frag(&qb[(size_t)(m0 + nrow) * DK + ks * 32 + half * 8],
                       &qb[(size_t)(m0 + nrow) * DK + ks * 32 + 16 + half * 8]);

  v8f O[4] = {};
  float mrun[8], lrun[8];
#pragma unroll
  for (int r = 0; r < 8; ++r) { mrun[r] = -3.0e38f; lrun[r] = 0.0f; }

  const int sr = tid >> 3, sc8 = tid & 7;  // cooperative staging coords

  for (int key0 = 0; key0 < S; key0 += 32) {
    // stage K chunk row-major via async-to-LDS (ASYNCcnt path)
    async_copy16(&kb[(size_t)(key0 + sr) * DK + sc8 * 8], &lK[sr][sc8 * 8]);
    // stage V chunk transposed (needs per-element scatter)
    v8bf tv = *(const v8bf*)&vb[(size_t)(key0 + sr) * DK + sc8 * 8];
#pragma unroll
    for (int j = 0; j < 8; ++j) lV[sc8 * 8 + j][sr] = tv[j];
    if (key0 + 32 < S) {
      __builtin_prefetch(&kb[(size_t)(key0 + 32 + sr) * DK], 0, 1);
      __builtin_prefetch(&vb[(size_t)(key0 + 32 + sr) * DK], 0, 1);
    }
    wait_async();
    __syncthreads();

    // scores: 16 rows x 32 keys, contract dk=64 in two K=32 steps
    v8f sc[2] = {};
#pragma unroll
    for (int nt = 0; nt < 2; ++nt) {
#pragma unroll
      for (int ks = 0; ks < 2; ++ks) {
        v16bf bf = make_frag(&lK[nt * 16 + nrow][ks * 32 + half * 8],
                             &lK[nt * 16 + nrow][ks * 32 + 16 + half * 8]);
        sc[nt] = WMMA_BF16(qf[ks], bf, sc[nt]);
      }
    }

    // online softmax (row stats reduced across the 16-lane row group)
    float cmax[8];
#pragma unroll
    for (int r = 0; r < 8; ++r) {
      sc[0][r] *= 0.125f;  // 1/sqrt(64)
      sc[1][r] *= 0.125f;
      cmax[r] = fmaxf(sc[0][r], sc[1][r]);
    }
#pragma unroll
    for (int mask = 1; mask < 16; mask <<= 1)
#pragma unroll
      for (int r = 0; r < 8; ++r)
        cmax[r] = fmaxf(cmax[r], __shfl_xor(cmax[r], mask, 32));

    float corr[8], p0[8], p1[8], csum[8];
#pragma unroll
    for (int r = 0; r < 8; ++r) {
      float nm = fmaxf(mrun[r], cmax[r]);
      corr[r] = __expf(mrun[r] - nm);
      mrun[r] = nm;
      p0[r] = __expf(sc[0][r] - nm);
      p1[r] = __expf(sc[1][r] - nm);
      csum[r] = p0[r] + p1[r];
    }
#pragma unroll
    for (int mask = 1; mask < 16; mask <<= 1)
#pragma unroll
      for (int r = 0; r < 8; ++r) csum[r] += __shfl_xor(csum[r], mask, 32);
#pragma unroll
    for (int r = 0; r < 8; ++r) {
      lrun[r] = lrun[r] * corr[r] + csum[r];
#pragma unroll
      for (int ot = 0; ot < 4; ++ot) O[ot][r] *= corr[r];
    }

    // P: C layout -> LDS -> A layout (per-wave, s_wait_dscnt only)
#pragma unroll
    for (int r = 0; r < 8; ++r) {
      lP[wave][half * 8 + r][nrow] = (__bf16)p0[r];
      lP[wave][half * 8 + r][16 + nrow] = (__bf16)p1[r];
    }
    asm volatile("s_wait_dscnt 0" ::: "memory");
    v16bf pf = make_frag(&lP[wave][nrow][half * 8], &lP[wave][nrow][16 + half * 8]);

    // O += P(16x32) x V(32x64)
#pragma unroll
    for (int ot = 0; ot < 4; ++ot) {
      v16bf vf = make_frag(&lV[ot * 16 + nrow][half * 8],
                           &lV[ot * 16 + nrow][16 + half * 8]);
      O[ot] = WMMA_BF16(pf, vf, O[ot]);
    }
    __syncthreads();
  }

  // finalize: divide by row sums, store heads[b,s,h*64+dk] (bf16)
#pragma unroll
  for (int r = 0; r < 8; ++r) {
    float inv = 1.0f / lrun[r];
    int s = m0 + half * 8 + r;
#pragma unroll
    for (int ot = 0; ot < 4; ++ot)
      heads[((size_t)(bb * S + s)) * (H * DK) + h * DK + ot * 16 + nrow] =
          (__bf16)(O[ot][r] * inv);
  }
}

// ---------------------------------------------------------------------------
// Kernel 3: output projection + residual.  tmp = heads(bf16) @ W_o + Q
// Grid: (M/256, N/64), block 256.  A tile staged via async-to-LDS.
// ---------------------------------------------------------------------------
__global__ __launch_bounds__(256)
void out_proj_kernel(const __bf16* __restrict__ heads, const float* __restrict__ Wo,
                     const float* __restrict__ resid, float* __restrict__ tmp) {
  constexpr int DM = 1024;
  const int row0 = blockIdx.x * 256, n0 = blockIdx.y * 64;
  __shared__ __align__(16) __bf16 lA[256][40];
  __shared__ __align__(16) __bf16 lB[64][40];
  const int tid = threadIdx.x, lane = tid & 31, wave = tid >> 5;
  const int m0 = wave * 32, nrow = lane & 15, half = lane >> 4;
  v8f acc[2][4] = {};

  for (int kk = 0; kk < DM; kk += 32) {
#pragma unroll
    for (int t = 0; t < 4; ++t) {  // A: already bf16, async 16B copies
      int idx = tid + t * 256;
      int r = idx >> 2, c8 = idx & 3;
      async_copy16(&heads[(size_t)(row0 + r) * DM + kk + c8 * 8], &lA[r][c8 * 8]);
    }
#pragma unroll
    for (int t = 0; t < 2; ++t) {  // B: f32 -> bf16, transposed
      int idx = tid + t * 256;
      int r = idx >> 4, c4 = idx & 15;
      v4f d = *(const v4f*)&Wo[(size_t)(kk + r) * DM + n0 + c4 * 4];
#pragma unroll
      for (int j = 0; j < 4; ++j) lB[c4 * 4 + j][r] = (__bf16)d[j];
    }
    wait_async();
    __syncthreads();

    v16bf a[2];
#pragma unroll
    for (int mi = 0; mi < 2; ++mi)
      a[mi] = make_frag(&lA[m0 + mi * 16 + nrow][half * 8],
                        &lA[m0 + mi * 16 + nrow][16 + half * 8]);
#pragma unroll
    for (int nt = 0; nt < 4; ++nt) {
      v16bf b = make_frag(&lB[nt * 16 + nrow][half * 8],
                          &lB[nt * 16 + nrow][16 + half * 8]);
#pragma unroll
      for (int mi = 0; mi < 2; ++mi) acc[mi][nt] = WMMA_BF16(a[mi], b, acc[mi][nt]);
    }
    __syncthreads();
  }

#pragma unroll
  for (int mi = 0; mi < 2; ++mi)
#pragma unroll
    for (int nt = 0; nt < 4; ++nt)
#pragma unroll
      for (int r = 0; r < 8; ++r) {
        int gr = row0 + m0 + mi * 16 + half * 8 + r;
        int col = n0 + nt * 16 + nrow;
        tmp[(size_t)gr * DM + col] = acc[mi][nt][r] + resid[(size_t)gr * DM + col];
      }
}

// ---------------------------------------------------------------------------
// Kernel 4: LayerNorm over rows of 1024.  One row per 256-thread block.
// ---------------------------------------------------------------------------
__global__ __launch_bounds__(256)
void layernorm_kernel(const float* __restrict__ x, const float* __restrict__ gamma,
                      const float* __restrict__ beta, float* __restrict__ out) {
  constexpr int DM = 1024;
  const int row = blockIdx.x;
  const int tid = threadIdx.x, lane = tid & 31, wave = tid >> 5;
  const float* xr = x + (size_t)row * DM;
  float vals[4], s1 = 0.0f, s2 = 0.0f;
#pragma unroll
  for (int j = 0; j < 4; ++j) {
    vals[j] = xr[tid + j * 256];
    s1 += vals[j];
    s2 += vals[j] * vals[j];
  }
#pragma unroll
  for (int mask = 16; mask >= 1; mask >>= 1) {
    s1 += __shfl_xor(s1, mask, 32);
    s2 += __shfl_xor(s2, mask, 32);
  }
  __shared__ float red1[8], red2[8];
  if (lane == 0) { red1[wave] = s1; red2[wave] = s2; }
  __syncthreads();
  float t1 = 0.0f, t2 = 0.0f;
#pragma unroll
  for (int w = 0; w < 8; ++w) { t1 += red1[w]; t2 += red2[w]; }
  float mean = t1 * (1.0f / DM);
  float var = t2 * (1.0f / DM) - mean * mean;
  float rstd = rsqrtf(var + 1e-5f);
#pragma unroll
  for (int j = 0; j < 4; ++j) {
    int c = tid + j * 256;
    out[(size_t)row * DM + c] = gamma[c] * (vals[j] - mean) * rstd + beta[c];
  }
}

// ---------------------------------------------------------------------------
extern "C" void kernel_launch(void* const* d_in, const int* in_sizes, int n_in,
                              void* d_out, int out_size, void* d_ws, size_t ws_size,
                              hipStream_t stream) {
  (void)in_sizes; (void)n_in; (void)out_size; (void)ws_size;
  const float* Q = (const float*)d_in[0];
  const float* K = (const float*)d_in[1];
  const float* V = (const float*)d_in[2];
  const float* Wq = (const float*)d_in[3];
  const float* Wk = (const float*)d_in[4];
  const float* Wv = (const float*)d_in[5];
  const float* Wo = (const float*)d_in[6];
  const float* gamma = (const float*)d_in[7];
  const float* beta = (const float*)d_in[8];
  float* out = (float*)d_out;

  char* ws = (char*)d_ws;
  const size_t BF = (size_t)4 * 16 * 2048 * 64 * sizeof(__bf16);  // 16 MiB
  __bf16* qb = (__bf16*)(ws);
  __bf16* kb = (__bf16*)(ws + BF);
  __bf16* vb = (__bf16*)(ws + 2 * BF);
  __bf16* heads = (__bf16*)(ws + 3 * BF);
  float* tmp = (float*)(ws + 4 * BF);  // 32 MiB f32 pre-LN buffer

  dim3 blk(256);
  qkv_proj_kernel<<<dim3(32, 16, 3), blk, 0, stream>>>(Q, K, V, Wq, Wk, Wv, qb, kb, vb);
  attention_kernel<<<dim3(16, 64), blk, 0, stream>>>(qb, kb, vb, heads);
  out_proj_kernel<<<dim3(32, 16), blk, 0, stream>>>(heads, Wo, Q, tmp);
  layernorm_kernel<<<dim3(8192), blk, 0, stream>>>(tmp, gamma, beta, out);
}
